// RecurrentDecisionMaker_72627896975390
// MI455X (gfx1250) — matrix operand order repaired and checked by reading, not compiled, not run on hardware
//
#include <hip/hip_runtime.h>
#include <hip/hip_bf16.h>

typedef __attribute__((ext_vector_type(16))) __bf16 v16bf;
typedef __attribute__((ext_vector_type(8)))  __bf16 v8bf;
typedef __attribute__((ext_vector_type(8)))  float  v8f;

#define GRU_B   32
#define GRU_T   2048
#define GRU_F   128
#define GRU_U   256
#define GRU_N3  768      // 3*U
#define NFRAME  20

// ---------------------------------------------------------------------------
// Transpose + convert f32 weights [K][N] -> bf16 [N][K] so WMMA B-fragments
// are two contiguous 16B runs along K per lane.
// ---------------------------------------------------------------------------
__global__ void wt_transpose_bf16(const float* __restrict__ in,
                                  __bf16* __restrict__ out, int K, int N) {
  int idx = blockIdx.x * blockDim.x + threadIdx.x;
  if (idx >= K * N) return;
  int n = idx / K;
  int k = idx - n * K;
  out[idx] = (__bf16)in[(size_t)k * N + n];
}

// ---------------------------------------------------------------------------
// xw = A @ W  (no bias; bias folded into gate phase).
// A: [M][K] f32 (layer0 input) or bf16 (layer1 input = h0).
// W: bf16 transposed [N][K].  Out: bf16 [M][N].
// One wave per 16x16 tile, 8 waves/block, K-loop of 16x16x32 bf16 WMMA.
// ---------------------------------------------------------------------------
template <bool ABF16>
__launch_bounds__(256)
__global__ void gemm_xw_kernel(const void* __restrict__ Ap,
                               const __bf16* __restrict__ wt,
                               __bf16* __restrict__ out, int K, int N) {
  const int lane  = threadIdx.x & 31;
  const int wave  = threadIdx.x >> 5;
  const int lh    = lane & 15;
  const int hi    = lane >> 4;           // 0 = lanes 0-15, 1 = lanes 16-31
  const int mbase = blockIdx.x * 16;
  const int nbase = (blockIdx.y * 8 + wave) * 16;
  const int m = mbase + lh;
  const int n = nbase + lh;

  v8f c = {};
  for (int kc = 0; kc < K; kc += 32) {
    // A-fragment per ISA layout: lane<16 -> K kc..kc+7 & kc+16..kc+23,
    //                            lane>=16 -> K kc+8..kc+15 & kc+24..kc+31
    v16bf a;
    if (ABF16) {
      const __bf16* A = (const __bf16*)Ap;
      v8bf r0 = *(const v8bf*)(A + (size_t)m * K + kc + hi * 8);
      v8bf r1 = *(const v8bf*)(A + (size_t)m * K + kc + 16 + hi * 8);
#pragma unroll
      for (int i = 0; i < 8; ++i) { a[i] = r0[i]; a[i + 8] = r1[i]; }
    } else {
      const float* A  = (const float*)Ap;
      const float* p0 = A + (size_t)m * K + kc + hi * 8;
      const float* p1 = A + (size_t)m * K + kc + 16 + hi * 8;
#pragma unroll
      for (int i = 0; i < 8; ++i) { a[i] = (__bf16)p0[i]; a[i + 8] = (__bf16)p1[i]; }
    }
    // B-fragment: same K order as A (lane<16 -> kc..kc+7 & kc+16..kc+23, lane>=16 -> kc+8..kc+15 & kc+24..kc+31)
    v16bf b;
    {
      v8bf q0 = *(const v8bf*)(wt + (size_t)n * K + kc + hi * 8);
      v8bf q1 = *(const v8bf*)(wt + (size_t)n * K + kc + 16 + hi * 8);
#pragma unroll
      for (int i = 0; i < 8; ++i) { b[i] = q0[i]; b[i + 8] = q1[i]; }
    }
    c = __builtin_amdgcn_wmma_f32_16x16x32_bf16(false, a, false, b,
                                                (short)0, c, false, false);
  }
#pragma unroll
  for (int r = 0; r < 8; ++r)
    out[(size_t)(mbase + r + 8 * hi) * N + nbase + lh] = (__bf16)c[r];
}

// ---------------------------------------------------------------------------
// Sequential GRU scan, one workgroup per direction (grid.x = 2).
// 768 threads = 24 waves; wave w owns N-columns [32w, 32w+32) for both
// 16-row M-tiles.  Recurrent weights for the EVEN ntile of each wave are
// cached in LDS once (24 x 8KB = 192KB); the ODD ntile streams from L2 each
// step (weights stay L2-resident).  A per-step SGPR "laundered" zero offset
// stops the compiler from hoisting W-fragment loads across timesteps (which
// previously caused scratch spills in the critical sequential loop).
// LDS: rec bf16 48KB + h f32 32KB + h bf16 16KB + Wcache 192KB = 288KB.
// ---------------------------------------------------------------------------
__launch_bounds__(768)
__global__ void gru_scan_kernel(const __bf16* __restrict__ xw_all, // [2][B*T][768]
                                const __bf16* __restrict__ rk_f,   // [768][256] bf16
                                const __bf16* __restrict__ rk_b,
                                const float* __restrict__ bias_f,  // [2][768]
                                const float* __restrict__ bias_b,
                                __bf16* __restrict__ hout,         // [B][T][512]
                                int T) {
  constexpr int REC_BYTES = GRU_B * GRU_N3 * 2;   // 49152
  constexpr int HF_BYTES  = GRU_B * GRU_U * 4;    // 32768
  constexpr int HBF_BYTES = GRU_B * GRU_U * 2;    // 16384
  extern __shared__ char smem[];
  __bf16* recb = (__bf16*)smem;                                   // [32][768]
  float*  hf   = (float*)(smem + REC_BYTES);                      // [32][256]
  __bf16* hbf  = (__bf16*)(smem + REC_BYTES + HF_BYTES);          // [32][256]
  __bf16* wlds = (__bf16*)(smem + REC_BYTES + HF_BYTES + HBF_BYTES); // 24*16*256

  const int dir = blockIdx.x;
  const __bf16* xw   = xw_all + (size_t)dir * GRU_B * T * GRU_N3;
  const __bf16* rk   = dir ? rk_b : rk_f;
  const float*  bias = dir ? bias_b : bias_f;
  const float*  b0   = bias;
  const float*  b1   = bias + GRU_N3;
  const int colofs   = dir * GRU_U;

  const int tid  = threadIdx.x;
  const int lane = tid & 31;
  const int wave = tid >> 5;        // 0..23
  const int lh   = lane & 15;
  const int hi   = lane >> 4;

  // init hidden state
  for (int i = tid; i < GRU_B * GRU_U; i += 768) {
    hf[i]  = 0.f;
    hbf[i] = (__bf16)0.f;
  }
  // cache this wave's even-ntile weight block (16 cols x 256 K) into LDS
  {
    const v8bf* src = (const v8bf*)(rk + (size_t)(2 * wave) * 16 * GRU_U);
    v8bf*       dst = (v8bf*)(wlds + (size_t)wave * 16 * GRU_U);
#pragma unroll 4
    for (int i = lane; i < 16 * GRU_U / 8; i += 32) dst[i] = src[i];
  }
  __syncthreads();

  const int nb0 = (2 * wave) * 16;      // even ntile (LDS-cached)
  const int nb1 = (2 * wave + 1) * 16;  // odd ntile  (L2-streamed)
  const __bf16* wl = wlds + (size_t)(wave * 16 + lh) * GRU_U + hi * 8;
  const __bf16* wg = rk + (size_t)(nb1 + lh) * GRU_U + hi * 8;
  const __bf16* a_base0 = hbf + (size_t)lh * GRU_U;         // M-tile 0
  const __bf16* a_base1 = hbf + (size_t)(16 + lh) * GRU_U;  // M-tile 1

  for (int t = 0; t < T; ++t) {
    const int tt = dir ? (T - 1 - t) : t;

    // Laundered zero: redefined by volatile asm each step so W-fragment
    // addresses are not loop-invariant -> no hoist/spill across timesteps.
    int tz = 0;
    asm volatile("" : "+s"(tz));

    v8f c00 = {}, c01 = {}, c10 = {}, c11 = {};
#pragma unroll
    for (int kc = 0; kc < GRU_U; kc += 32) {
      v16bf a0, a1;
      {
        v8bf r0 = *(const v8bf*)(a_base0 + kc + hi * 8);
        v8bf r1 = *(const v8bf*)(a_base0 + kc + 16 + hi * 8);
        v8bf s0 = *(const v8bf*)(a_base1 + kc + hi * 8);
        v8bf s1 = *(const v8bf*)(a_base1 + kc + 16 + hi * 8);
#pragma unroll
        for (int i = 0; i < 8; ++i) {
          a0[i] = r0[i]; a0[i + 8] = r1[i];
          a1[i] = s0[i]; a1[i + 8] = s1[i];
        }
      }
      v16bf bw0, bw1;
      {
        v8bf p0 = *(const v8bf*)(wl + kc + tz);      // LDS-cached W
        v8bf p1 = *(const v8bf*)(wl + kc + 16 + tz);
#pragma unroll
        for (int i = 0; i < 8; ++i) { bw0[i] = p0[i]; bw0[i + 8] = p1[i]; }
      }
      c00 = __builtin_amdgcn_wmma_f32_16x16x32_bf16(false, a0, false, bw0, (short)0, c00, false, false);
      c10 = __builtin_amdgcn_wmma_f32_16x16x32_bf16(false, a1, false, bw0, (short)0, c10, false, false);
      {
        v8bf g0 = *(const v8bf*)(wg + kc + tz);      // L2-streamed W
        v8bf g1 = *(const v8bf*)(wg + kc + 16 + tz);
#pragma unroll
        for (int i = 0; i < 8; ++i) { bw1[i] = g0[i]; bw1[i + 8] = g1[i]; }
      }
      c01 = __builtin_amdgcn_wmma_f32_16x16x32_bf16(false, a0, false, bw1, (short)0, c01, false, false);
      c11 = __builtin_amdgcn_wmma_f32_16x16x32_bf16(false, a1, false, bw1, (short)0, c11, false, false);
    }
#pragma unroll
    for (int r = 0; r < 8; ++r) {
      const int mr = r + 8 * hi;
      recb[(size_t)mr * GRU_N3 + nb0 + lh]        = (__bf16)c00[r];
      recb[(size_t)mr * GRU_N3 + nb1 + lh]        = (__bf16)c01[r];
      recb[(size_t)(16 + mr) * GRU_N3 + nb0 + lh] = (__bf16)c10[r];
      recb[(size_t)(16 + mr) * GRU_N3 + nb1 + lh] = (__bf16)c11[r];
    }
    __syncthreads();

    // Fused gate update over all 32*256 hidden elements.
    for (int idx = tid; idx < GRU_B * GRU_U; idx += 768) {
      const int b = idx >> 8;
      const int u = idx & (GRU_U - 1);
      const size_t xrow = ((size_t)b * T + tt) * GRU_N3;
      float xz = (float)xw[xrow + u]             + b0[u];
      float xr = (float)xw[xrow + GRU_U + u]     + b0[GRU_U + u];
      float xh = (float)xw[xrow + 2 * GRU_U + u] + b0[2 * GRU_U + u];
      float rz = (float)recb[(size_t)b * GRU_N3 + u]             + b1[u];
      float rr = (float)recb[(size_t)b * GRU_N3 + GRU_U + u]     + b1[GRU_U + u];
      float rh = (float)recb[(size_t)b * GRU_N3 + 2 * GRU_U + u] + b1[2 * GRU_U + u];
      float z  = 1.f / (1.f + __expf(-(xz + rz)));
      float r  = 1.f / (1.f + __expf(-(xr + rr)));
      float hh = fmaxf(0.f, xh + r * rh);
      float hn = z * hf[idx] + (1.f - z) * hh;
      hf[idx]  = hn;
      hbf[idx] = (__bf16)hn;
      hout[((size_t)b * T + tt) * (2 * GRU_U) + colofs + u] = (__bf16)hn;
    }
    __syncthreads();
  }
}

// ---------------------------------------------------------------------------
// Dense head: logits = h1 @ Wd + bd, softmax over 20 classes. 1.3 GFLOP.
// ---------------------------------------------------------------------------
__launch_bounds__(256)
__global__ void head_kernel(const __bf16* __restrict__ h1, // [rows][512]
                            const float* __restrict__ Wd,  // [512][20]
                            const float* __restrict__ bd,  // [20]
                            float* __restrict__ out, int rows) {
  int gid = blockIdx.x * blockDim.x + threadIdx.x;
  if (gid >= rows) return;
  float acc[NFRAME];
#pragma unroll
  for (int k = 0; k < NFRAME; ++k) acc[k] = bd[k];
  const __bf16* hrow = h1 + (size_t)gid * (2 * GRU_U);
  for (int d = 0; d < 2 * GRU_U; ++d) {
    float hv = (float)hrow[d];
#pragma unroll
    for (int k = 0; k < NFRAME; ++k) acc[k] = fmaf(hv, Wd[d * NFRAME + k], acc[k]);
  }
  float mx = acc[0];
#pragma unroll
  for (int k = 1; k < NFRAME; ++k) mx = fmaxf(mx, acc[k]);
  float s = 0.f;
#pragma unroll
  for (int k = 0; k < NFRAME; ++k) { acc[k] = __expf(acc[k] - mx); s += acc[k]; }
  float inv = 1.f / s;
#pragma unroll
  for (int k = 0; k < NFRAME; ++k) out[(size_t)gid * NFRAME + k] = acc[k] * inv;
}

// ---------------------------------------------------------------------------
extern "C" void kernel_launch(void* const* d_in, const int* in_sizes, int n_in,
                              void* d_out, int out_size, void* d_ws, size_t ws_size,
                              hipStream_t stream) {
  const float* x    = (const float*)d_in[0];
  const float* k0f  = (const float*)d_in[1];
  const float* rk0f = (const float*)d_in[2];
  const float* b0f  = (const float*)d_in[3];
  const float* k0b  = (const float*)d_in[4];
  const float* rk0b = (const float*)d_in[5];
  const float* b0b  = (const float*)d_in[6];
  const float* k1f  = (const float*)d_in[7];
  const float* rk1f = (const float*)d_in[8];
  const float* b1f  = (const float*)d_in[9];
  const float* k1b  = (const float*)d_in[10];
  const float* rk1b = (const float*)d_in[11];
  const float* b1b  = (const float*)d_in[12];
  const float* Wd   = (const float*)d_in[13];
  const float* bd   = (const float*)d_in[14];
  (void)in_sizes; (void)n_in; (void)out_size; (void)ws_size;

  char* ws = (char*)d_ws;
  size_t off = 0;
  auto alloc = [&](size_t bytes) -> char* {
    char* p = ws + off;
    off += (bytes + 255) & ~(size_t)255;
    return p;
  };

  const int ROWS = GRU_B * GRU_T;  // 65536

  __bf16* k0f_t  = (__bf16*)alloc((size_t)GRU_N3 * GRU_F * 2);
  __bf16* k0b_t  = (__bf16*)alloc((size_t)GRU_N3 * GRU_F * 2);
  __bf16* rk0f_t = (__bf16*)alloc((size_t)GRU_N3 * GRU_U * 2);
  __bf16* rk0b_t = (__bf16*)alloc((size_t)GRU_N3 * GRU_U * 2);
  __bf16* rk1f_t = (__bf16*)alloc((size_t)GRU_N3 * GRU_U * 2);
  __bf16* rk1b_t = (__bf16*)alloc((size_t)GRU_N3 * GRU_U * 2);
  __bf16* k1f_t  = (__bf16*)alloc((size_t)GRU_N3 * (2 * GRU_U) * 2);
  __bf16* k1b_t  = (__bf16*)alloc((size_t)GRU_N3 * (2 * GRU_U) * 2);
  __bf16* xw     = (__bf16*)alloc((size_t)2 * ROWS * GRU_N3 * 2);   // both dirs
  __bf16* h0     = (__bf16*)alloc((size_t)ROWS * (2 * GRU_U) * 2);
  __bf16* h1     = (__bf16*)alloc((size_t)ROWS * (2 * GRU_U) * 2);

  // 1) weight transpose/convert to bf16 [N][K]
  {
    const int thr = 256;
    int n_k0 = GRU_N3 * GRU_F;        // 98304
    int n_rk = GRU_N3 * GRU_U;        // 196608
    int n_k1 = GRU_N3 * (2 * GRU_U);  // 393216
    wt_transpose_bf16<<<(n_k0 + thr - 1) / thr, thr, 0, stream>>>(k0f, k0f_t, GRU_F, GRU_N3);
    wt_transpose_bf16<<<(n_k0 + thr - 1) / thr, thr, 0, stream>>>(k0b, k0b_t, GRU_F, GRU_N3);
    wt_transpose_bf16<<<(n_rk + thr - 1) / thr, thr, 0, stream>>>(rk0f, rk0f_t, GRU_U, GRU_N3);
    wt_transpose_bf16<<<(n_rk + thr - 1) / thr, thr, 0, stream>>>(rk0b, rk0b_t, GRU_U, GRU_N3);
    wt_transpose_bf16<<<(n_rk + thr - 1) / thr, thr, 0, stream>>>(rk1f, rk1f_t, GRU_U, GRU_N3);
    wt_transpose_bf16<<<(n_rk + thr - 1) / thr, thr, 0, stream>>>(rk1b, rk1b_t, GRU_U, GRU_N3);
    wt_transpose_bf16<<<(n_k1 + thr - 1) / thr, thr, 0, stream>>>(k1f, k1f_t, 2 * GRU_U, GRU_N3);
    wt_transpose_bf16<<<(n_k1 + thr - 1) / thr, thr, 0, stream>>>(k1b, k1b_t, 2 * GRU_U, GRU_N3);
  }

  dim3 gGemm(ROWS / 16, GRU_N3 / (16 * 8));  // (4096, 6)
  const size_t smem_bytes = (size_t)GRU_B * GRU_N3 * 2   // rec bf16   48KB
                          + (size_t)GRU_B * GRU_U * 4    // h f32      32KB
                          + (size_t)GRU_B * GRU_U * 2    // h bf16     16KB
                          + (size_t)24 * 16 * GRU_U * 2; // W cache   192KB  => 288KB
  const size_t dirstride = (size_t)ROWS * GRU_N3;

  // 2) layer-0 input projections (f32 A, K=128)
  gemm_xw_kernel<false><<<gGemm, 256, 0, stream>>>(x, k0f_t, xw,             GRU_F, GRU_N3);
  gemm_xw_kernel<false><<<gGemm, 256, 0, stream>>>(x, k0b_t, xw + dirstride, GRU_F, GRU_N3);

  // 3) layer-0 bidirectional scan -> h0 [B][T][512]
  gru_scan_kernel<<<2, 768, smem_bytes, stream>>>(xw, rk0f_t, rk0b_t, b0f, b0b, h0, GRU_T);

  // 4) layer-1 input projections (bf16 A = h0, K=512)
  gemm_xw_kernel<true><<<gGemm, 256, 0, stream>>>(h0, k1f_t, xw,             2 * GRU_U, GRU_N3);
  gemm_xw_kernel<true><<<gGemm, 256, 0, stream>>>(h0, k1b_t, xw + dirstride, 2 * GRU_U, GRU_N3);

  // 5) layer-1 bidirectional scan -> h1 [B][T][512]
  gru_scan_kernel<<<2, 768, smem_bytes, stream>>>(xw, rk1f_t, rk1b_t, b1f, b1b, h1, GRU_T);

  // 6) dense + softmax head -> d_out [B][T][20] f32
  head_kernel<<<(ROWS + 255) / 256, 256, 0, stream>>>(h1, Wd, bd, (float*)d_out, ROWS);
}
